// Mamba4Rec_23965917511743
// MI455X (gfx1250) — compile-verified, matrix operand order checked by reading, not run on hardware
//
#include <hip/hip_runtime.h>
#include <hip/hip_bf16.h>

typedef __attribute__((ext_vector_type(16))) _Float16 v16h;
typedef __attribute__((ext_vector_type(8)))  _Float16 v8h;
typedef __attribute__((ext_vector_type(8)))  float    v8f;

typedef __attribute__((address_space(1))) int gas_i32;   // global (device) dword
typedef __attribute__((address_space(3))) int lds_i32;   // LDS dword

#if __has_builtin(__builtin_amdgcn_global_load_async_to_lds_b32) && \
    __has_builtin(__builtin_amdgcn_s_wait_asynccnt)
#define HAVE_ASYNC_LDS 1
#endif

static constexpr int  kBatch  = 512;
static constexpr int  kSeq    = 200;
static constexpr int  kDModel = 128;
static constexpr int  kDInner = 256;
static constexpr int  kDState = 32;
static constexpr int  kDtRank = 8;
static constexpr int  kXdLd   = 80;      // x_dbl row stride, padded 72 -> 80
static constexpr long long kM = (long long)kBatch * kSeq;   // 102400 token rows

// ---------------------------------------------------------------- helpers
__device__ __forceinline__ float silu_f(float x)  { return x / (1.0f + __expf(-x)); }
__device__ __forceinline__ float gelu_f(float x)  { return 0.5f * x * (1.0f + erff(x * 0.70710678118654752f)); }
__device__ __forceinline__ float softplus_f(float x) { return (x > 20.0f) ? x : log1pf(__expf(x)); }

__device__ __forceinline__ float wave_sum(float v) {
#pragma unroll
  for (int off = 16; off; off >>= 1) v += __shfl_xor(v, off, 32);
  return v;
}

__device__ __forceinline__ v16h cat16(v8h lo, v8h hi) {
  return __builtin_shufflevector(lo, hi, 0, 1, 2, 3, 4, 5, 6, 7,
                                 8, 9, 10, 11, 12, 13, 14, 15);
}

// ---------------------------------------------------------------- WMMA GEMM (guard-free)
// C[M,N] = act(A[M,K] * W[N,K]^T + bias).  Requirements (enforced by padding):
//   M % 512 == 0, N % 16 == 0, K % 32 == 0, K <= 512, lda/ldw % 8 == 0.
// Block = 256 threads = 8 waves; wave owns 4 M-tiles x 1 N-tile (64x16 out).
// 16-column weight slab staged once into LDS (async on CDNA5) and shared by
// all 8 waves.  Fragment layout per CDNA5 ISA §7.12.2.
__global__ __launch_bounds__(256) void wmma_gemm_kernel(
    const _Float16* __restrict__ A, const _Float16* __restrict__ W,
    const float* __restrict__ bias,
    float* __restrict__ C, _Float16* __restrict__ C16,
    int N, int K, int lda, int ldw, int ldc, int act) {
  __shared__ __align__(16) _Float16 ldsW[16 * 512];

  const int tid = threadIdx.x;
  const int wave = tid >> 5, lane = tid & 31;
  const int rsel = lane & 15, half = lane >> 4;
  const int cols = blockIdx.y * 16;

#if defined(HAVE_ASYNC_LDS)
  // Async stage: W[cols .. cols+15, 0:K] -> LDS, one b32 per dword.
  const int dwPerRow = K >> 1;              // dwords per 16-column row
  for (int t = tid; t < 16 * dwPerRow; t += 256) {
    const int r = t / dwPerRow;
    const int c = (t - r * dwPerRow) << 1;  // f16 column, even
    gas_i32* src = (gas_i32*)(unsigned long long)(size_t)(W + (size_t)(cols + r) * ldw + c);
    lds_i32* dst = (lds_i32*)(unsigned long long)(size_t)(ldsW + r * K + c);
    __builtin_amdgcn_global_load_async_to_lds_b32(src, dst, 0, 0);
  }
  __builtin_amdgcn_s_wait_asynccnt(0);
  __syncthreads();
#else
  const int kv = K >> 3;                    // v8h chunks per row
  for (int t = tid; t < 16 * kv; t += 256) {
    const int r = t / kv;
    const int c = (t - r * kv) << 3;
    *(v8h*)(ldsW + r * K + c) = *(const v8h*)(W + (size_t)(cols + r) * ldw + c);
  }
  __syncthreads();
#endif

  const long long rowBase = ((long long)blockIdx.x * 8 + wave) * 64;
  const _Float16* aptr[4];
#pragma unroll
  for (int t = 0; t < 4; ++t)
    aptr[t] = A + (size_t)(rowBase + t * 16 + rsel) * lda;
  const _Float16* wrow = ldsW + rsel * K;

  v8f acc[4] = {};
  for (int kb = 0; kb < K; kb += 32) {
    const int k1 = kb + half * 8, k2 = k1 + 16;
    __builtin_prefetch(aptr[0] + k1 + 64, 0, 1);   // gfx1250 global_prefetch_b8
    const v16h bf = cat16(*(const v8h*)(wrow + k1), *(const v8h*)(wrow + k2));
#pragma unroll
    for (int t = 0; t < 4; ++t) {
      const v16h af = cat16(*(const v8h*)(aptr[t] + k1), *(const v8h*)(aptr[t] + k2));
      acc[t] = __builtin_amdgcn_wmma_f32_16x16x32_f16(false, af, false, bf,
                                                      (short)0, acc[t], false, false);
    }
  }

  const int col = cols + rsel;
  const float bv = bias ? bias[col] : 0.0f;
#pragma unroll
  for (int t = 0; t < 4; ++t) {
#pragma unroll
    for (int e = 0; e < 8; ++e) {
      const long long r = rowBase + t * 16 + e + half * 8;
      float v = acc[t][e] + bv;
      if (act == 1) v = gelu_f(v);
      else if (act == 2) v = softplus_f(v);
      if (C)   C[(size_t)r * ldc + col] = v;
      if (C16) C16[(size_t)r * ldc + col] = (_Float16)v;
    }
  }
}

// ---------------------------------------------------------------- f32 -> f16 (plain + padded)
__global__ void f2h_kernel(const float* __restrict__ src, _Float16* __restrict__ dst, long long n) {
  long long i = (long long)blockIdx.x * blockDim.x + threadIdx.x;
  if (i < n) dst[i] = (_Float16)src[i];
}

__global__ void f2h_pad_kernel(const float* __restrict__ src, _Float16* __restrict__ dst,
                               int Ns, int Ks, int Np, int Kp) {
  long long i = (long long)blockIdx.x * blockDim.x + threadIdx.x;
  if (i >= (long long)Np * Kp) return;
  const int r = (int)(i / Kp), c = (int)(i % Kp);
  dst[i] = (r < Ns && c < Ks) ? (_Float16)src[(size_t)r * Ks + c] : (_Float16)0.0f;
}

// ---------------------------------------------------------------- LayerNorm (+residual / +gather)
__global__ void ln_kernel(const float* __restrict__ a, const float* __restrict__ res,
                          const int* __restrict__ gidx, const float* __restrict__ emb,
                          const float* __restrict__ w, const float* __restrict__ bias,
                          float* __restrict__ out32, _Float16* __restrict__ out16, long long M) {
  const int wave = threadIdx.x >> 5, lane = threadIdx.x & 31;
  const long long row = (long long)blockIdx.x * 8 + wave;
  if (row >= M) return;
  const float* src = gidx ? (emb + (size_t)gidx[row] * kDModel) : (a + (size_t)row * kDModel);
  const float* r2  = res ? (res + (size_t)row * kDModel) : nullptr;
  float x[4], s = 0.0f, sq = 0.0f;
#pragma unroll
  for (int i = 0; i < 4; ++i) {
    const int c = lane + i * 32;
    x[i] = src[c] + (r2 ? r2[c] : 0.0f);
    s += x[i]; sq += x[i] * x[i];
  }
  s = wave_sum(s); sq = wave_sum(sq);
  const float mean = s * (1.0f / kDModel);
  float var = sq * (1.0f / kDModel) - mean * mean;
  var = fmaxf(var, 0.0f);
  const float inv = rsqrtf(var + 1e-12f);
#pragma unroll
  for (int i = 0; i < 4; ++i) {
    const int c = lane + i * 32;
    const float y = (x[i] - mean) * inv * w[c] + bias[c];
    out32[(size_t)row * kDModel + c] = y;
    if (out16) out16[(size_t)row * kDModel + c] = (_Float16)y;
  }
}

// ---------------------------------------------------------------- causal depthwise conv4 + SiLU
__global__ void conv_silu_kernel(const float* __restrict__ xz,
                                 const float* __restrict__ cw, const float* __restrict__ cb,
                                 float* __restrict__ xc, _Float16* __restrict__ xc16, long long n) {
  long long idx = (long long)blockIdx.x * blockDim.x + threadIdx.x;
  if (idx >= n) return;
  const long long m = idx >> 8;
  const int d = (int)(idx & 255);
  const int l = (int)(m % kSeq);
  float s = cb[d];
#pragma unroll
  for (int j = 0; j < 4; ++j) {
    const int lj = l - 3 + j;
    if (lj >= 0) s += cw[d * 4 + j] * xz[(size_t)(m - 3 + j) * (2 * kDInner) + d];
  }
  s = silu_f(s);
  xc[idx] = s;
  xc16[idx] = (_Float16)s;
}

// ---------------------------------------------------------------- selective scan
// One block per batch element; thread = channel d; 32-float state in registers.
// CDNA5 path: B_t/C_t double-buffered via global_load_async_to_lds_b32 so the
// next step's 256B fetch overlaps the current step's exp/FMA chain.
__global__ void scan_kernel(const float* __restrict__ delta, const float* __restrict__ u,
                            const float* __restrict__ xdbl, const float* __restrict__ A_log,
                            float* __restrict__ ybuf) {
  const int b = blockIdx.x, d = threadIdx.x;
  float Arow[kDState], h[kDState];
#pragma unroll
  for (int n = 0; n < kDState; ++n) {
    Arow[n] = -__expf(A_log[(size_t)d * kDState + n]);
    h[n] = 0.0f;
  }

#if defined(HAVE_ASYNC_LDS)
  __shared__ float sBC[2][2 * kDState];
  auto issue = [&](int l) {
    if (d < 2 * kDState) {
      const size_t m = (size_t)b * kSeq + l;
      gas_i32* src = (gas_i32*)(unsigned long long)(size_t)(xdbl + m * kXdLd + kDtRank + d);
      lds_i32* dst = (lds_i32*)(unsigned long long)(size_t)&sBC[l & 1][d];
      __builtin_amdgcn_global_load_async_to_lds_b32(src, dst, 0, 0);
    }
  };
  issue(0);
  for (int l = 0; l < kSeq; ++l) {
    if (l + 1 < kSeq) {
      issue(l + 1);                              // newest in flight
      __builtin_amdgcn_s_wait_asynccnt(1);       // oldest (step l) has landed
    } else {
      __builtin_amdgcn_s_wait_asynccnt(0);
    }
    __syncthreads();
    const float* bc = sBC[l & 1];
    const long long m = (long long)b * kSeq + l;
    const float dl = delta[(size_t)m * kDInner + d];
    const float du = dl * u[(size_t)m * kDInner + d];
    float y = 0.0f;
#pragma unroll
    for (int n = 0; n < kDState; ++n) {
      h[n] = __expf(dl * Arow[n]) * h[n] + du * bc[n];
      y += h[n] * bc[kDState + n];
    }
    ybuf[(size_t)m * kDInner + d] = y;
    __syncthreads();                             // reads done before buf reuse
  }
#else
  __shared__ float sBC[2 * kDState];
  for (int l = 0; l < kSeq; ++l) {
    const long long m = (long long)b * kSeq + l;
    if (d < 2 * kDState) sBC[d] = xdbl[(size_t)m * kXdLd + kDtRank + d];
    __syncthreads();
    const float dl = delta[(size_t)m * kDInner + d];
    const float du = dl * u[(size_t)m * kDInner + d];
    float y = 0.0f;
#pragma unroll
    for (int n = 0; n < kDState; ++n) {
      h[n] = __expf(dl * Arow[n]) * h[n] + du * sBC[n];
      y += h[n] * sBC[kDState + n];
    }
    ybuf[(size_t)m * kDInner + d] = y;
    __syncthreads();
  }
#endif
}

// ---------------------------------------------------------------- gate: (y + xi*D) * silu(z)
__global__ void gate_kernel(const float* __restrict__ ybuf, const float* __restrict__ xc,
                            const float* __restrict__ xz, const float* __restrict__ Dp,
                            _Float16* __restrict__ g16, long long n) {
  long long idx = (long long)blockIdx.x * blockDim.x + threadIdx.x;
  if (idx >= n) return;
  const long long m = idx >> 8;
  const int d = (int)(idx & 255);
  const float y = ybuf[idx] + xc[idx] * Dp[d];
  const float z = xz[(size_t)m * (2 * kDInner) + kDInner + d];
  g16[idx] = (_Float16)(y * silu_f(z));
}

// ---------------------------------------------------------------- final gather + dot
__global__ void final_kernel(const float* __restrict__ h, const int* __restrict__ seq,
                             const int* __restrict__ tgt, const float* __restrict__ emb,
                             float* __restrict__ out) {
  const int wave = threadIdx.x >> 5, lane = threadIdx.x & 31;
  const int b = blockIdx.x * 8 + wave;
  if (b >= kBatch) return;
  float cnt = 0.0f;
  for (int l = lane; l < kSeq; l += 32) cnt += (seq[b * kSeq + l] != 0) ? 1.0f : 0.0f;
  cnt = wave_sum(cnt);
  const int last = max((int)cnt - 1, 0);
  const float* hv = h + ((size_t)b * kSeq + last) * kDModel;
  const float* tv = emb + (size_t)tgt[b] * kDModel;
  float acc = 0.0f;
#pragma unroll
  for (int i = 0; i < 4; ++i) {
    const int c = lane + i * 32;
    acc += hv[c] * tv[c];
  }
  acc = wave_sum(acc);
  if (lane == 0) out[b] = acc;
}

// ---------------------------------------------------------------- TDM probe (never launched)
#if __has_builtin(__builtin_amdgcn_tensor_load_to_lds)
typedef __attribute__((ext_vector_type(4))) unsigned int v4u;
typedef __attribute__((ext_vector_type(8))) int v8i;
typedef __attribute__((ext_vector_type(4))) int v4i;
__global__ void cdna5_tdm_probe() {
  v4u g0 = {}; v8i g1 = {}; v4i g2 = {}; v4i g3 = {}; v8i g4 = {};
  __builtin_amdgcn_tensor_load_to_lds(g0, g1, g2, g3, g4, 0);  // 6-arg clang-23 form
#if __has_builtin(__builtin_amdgcn_s_wait_tensorcnt)
  __builtin_amdgcn_s_wait_tensorcnt(0);
#endif
}
#endif

// ================================================================ launcher
extern "C" void kernel_launch(void* const* d_in, const int* in_sizes, int n_in,
                              void* d_out, int out_size, void* d_ws, size_t ws_size,
                              hipStream_t stream) {
  const int*   item_seq  = (const int*)d_in[0];
  const int*   target_id = (const int*)d_in[1];
  const float* emb       = (const float*)d_in[2];
  const float* ln_w      = (const float*)d_in[3];
  const float* ln_b      = (const float*)d_in[4];
  const float* in_proj_w = (const float*)d_in[5];
  const float* conv_w    = (const float*)d_in[6];
  const float* conv_b    = (const float*)d_in[7];
  const float* x_proj_w  = (const float*)d_in[8];
  const float* dt_proj_w = (const float*)d_in[9];
  const float* dt_proj_b = (const float*)d_in[10];
  const float* A_log     = (const float*)d_in[11];
  const float* D_param   = (const float*)d_in[12];
  const float* out_proj_w= (const float*)d_in[13];
  const float* mln_w     = (const float*)d_in[14];
  const float* mln_b     = (const float*)d_in[15];
  const float* w1        = (const float*)d_in[16];
  const float* b1        = (const float*)d_in[17];
  const float* w2        = (const float*)d_in[18];
  const float* b2        = (const float*)d_in[19];
  const float* fln_w     = (const float*)d_in[20];
  const float* fln_b     = (const float*)d_in[21];
  float* out = (float*)d_out;

  char* ws = (char*)d_ws;
  size_t off = 0;
  auto alloc = [&](size_t bytes) -> char* {
    char* p = ws + off;
    off = (off + bytes + 255) & ~(size_t)255;
    return p;
  };
  const size_t M = (size_t)kM;

  _Float16* w16_in  = (_Float16*)alloc(2 * 512 * 128 * sizeof(_Float16));
  _Float16* w16_xp  = (_Float16*)alloc(2 * 80 * 256 * sizeof(_Float16));   // N padded 72->80
  _Float16* w16_dt  = (_Float16*)alloc(2 * 256 * 32 * sizeof(_Float16));   // K padded 8->32
  _Float16* w16_op  = (_Float16*)alloc(2 * 128 * 256 * sizeof(_Float16));
  _Float16* w16_w1  = (_Float16*)alloc(2 * 512 * 128 * sizeof(_Float16));
  _Float16* w16_w2  = (_Float16*)alloc(2 * 128 * 512 * sizeof(_Float16));

  float*    buf_h    = (float*)   alloc(M * 128 * 4);
  _Float16* buf_h16  = (_Float16*)alloc(M * 128 * 2);
  float*    buf_xz   = (float*)   alloc(M * 512 * 4);
  float*    buf_xc   = (float*)   alloc(M * 256 * 4);
  _Float16* buf_xc16 = (_Float16*)alloc(M * 256 * 2);
  float*    buf_xd   = (float*)   alloc(M * kXdLd * 4);
  _Float16* buf_xd16 = (_Float16*)alloc(M * kXdLd * 2);
  float*    buf_dl   = (float*)   alloc(M * 256 * 4);
  float*    buf_y    = (float*)   alloc(M * 256 * 4);
  _Float16* buf_g16  = (_Float16*)alloc(M * 256 * 2);
  float*    buf_m    = (float*)   alloc(M * 128 * 4);
  float*    buf_h2   = (float*)   alloc(M * 128 * 4);
  _Float16* buf_h216 = (_Float16*)alloc(M * 128 * 2);
  _Float16* buf_ff16 = (_Float16*)alloc(M * 512 * 2);

  auto f2h = [&](const float* s, _Float16* d, long long n) {
    f2h_kernel<<<(unsigned)((n + 255) / 256), 256, 0, stream>>>(s, d, n);
  };
  auto f2h_pad = [&](const float* s, _Float16* d, int Ns, int Ks, int Np, int Kp) {
    long long n = (long long)Np * Kp;
    f2h_pad_kernel<<<(unsigned)((n + 255) / 256), 256, 0, stream>>>(s, d, Ns, Ks, Np, Kp);
  };
  f2h(in_proj_w,  w16_in, 2LL * 512 * 128);
  f2h(out_proj_w, w16_op, 2LL * 128 * 256);
  f2h(w1,         w16_w1, 2LL * 512 * 128);
  f2h(w2,         w16_w2, 2LL * 128 * 512);
  for (int i = 0; i < 2; ++i) {
    f2h_pad(x_proj_w  + (size_t)i * 72 * 256, w16_xp + (size_t)i * 80 * 256, 72, 256, 80, 256);
    f2h_pad(dt_proj_w + (size_t)i * 256 * 8,  w16_dt + (size_t)i * 256 * 32, 256, 8, 256, 32);
  }

  auto gemm = [&](const _Float16* A, const _Float16* W, const float* bias,
                  float* C, _Float16* C16, int N, int K, int lda, int ldw, int ldc, int act) {
    dim3 grid((unsigned)(kM / 512), (unsigned)(N / 16));
    wmma_gemm_kernel<<<grid, 256, 0, stream>>>(A, W, bias, C, C16, N, K, lda, ldw, ldc, act);
  };

  const unsigned lnGrid = (unsigned)((kM + 7) / 8);
  const long long nInner = kM * 256;
  const unsigned ewGrid = (unsigned)((nInner + 255) / 256);

  // h = LN(emb[item_seq])
  ln_kernel<<<lnGrid, 256, 0, stream>>>(nullptr, nullptr, item_seq, emb,
                                        ln_w, ln_b, buf_h, buf_h16, kM);

  for (int i = 0; i < 2; ++i) {
    // in_proj: [M,128] x [512,128]^T -> xz [M,512]
    gemm(buf_h16, w16_in + (size_t)i * 512 * 128, nullptr,
         buf_xz, nullptr, 512, 128, 128, 128, 512, 0);
    // causal depthwise conv + SiLU
    conv_silu_kernel<<<ewGrid, 256, 0, stream>>>(buf_xz, conv_w + (size_t)i * 256 * 4,
                                                 conv_b + (size_t)i * 256,
                                                 buf_xc, buf_xc16, nInner);
    // x_proj: [M,256] x [80,256]^T -> x_dbl [M,80] (cols 72..79 zero pad)
    gemm(buf_xc16, w16_xp + (size_t)i * 80 * 256, nullptr,
         buf_xd, buf_xd16, 80, 256, 256, 256, kXdLd, 0);
    // dt_proj + softplus: K padded to 32 (weight cols 8..31 zero) -> delta [M,256]
    gemm(buf_xd16, w16_dt + (size_t)i * 256 * 32, dt_proj_b + (size_t)i * 256,
         buf_dl, nullptr, 256, 32, kXdLd, 32, 256, 2);
    // selective scan
    scan_kernel<<<kBatch, kDInner, 0, stream>>>(buf_dl, buf_xc, buf_xd,
                                                A_log + (size_t)i * 256 * 32, buf_y);
    // gating
    gate_kernel<<<ewGrid, 256, 0, stream>>>(buf_y, buf_xc, buf_xz,
                                            D_param + (size_t)i * 256, buf_g16, nInner);
    // out_proj: [M,256] x [128,256]^T -> m [M,128]
    gemm(buf_g16, w16_op + (size_t)i * 128 * 256, nullptr,
         buf_m, nullptr, 128, 256, 256, 256, 128, 0);
    // h2 = LN(m + h)
    ln_kernel<<<lnGrid, 256, 0, stream>>>(buf_m, buf_h, nullptr, nullptr,
                                          mln_w + (size_t)i * 128, mln_b + (size_t)i * 128,
                                          buf_h2, buf_h216, kM);
    // FFN1 + GELU: [M,128] x [512,128]^T -> [M,512] (f16 only)
    gemm(buf_h216, w16_w1 + (size_t)i * 512 * 128, b1 + (size_t)i * 512,
         nullptr, buf_ff16, 512, 128, 128, 128, 512, 1);
    // FFN2: [M,512] x [128,512]^T -> [M,128]
    gemm(buf_ff16, w16_w2 + (size_t)i * 128 * 512, b2 + (size_t)i * 128,
         buf_m, nullptr, 128, 512, 512, 512, 128, 0);
    // h = LN(ff + h2)
    ln_kernel<<<lnGrid, 256, 0, stream>>>(buf_m, buf_h2, nullptr, nullptr,
                                          fln_w + (size_t)i * 128, fln_b + (size_t)i * 128,
                                          buf_h, buf_h16, kM);
  }

  final_kernel<<<(kBatch + 7) / 8, 256, 0, stream>>>(buf_h, item_seq, target_id, emb, out);
  (void)in_sizes; (void)n_in; (void)out_size; (void)ws_size;
}
